// SentimentAnalyzer_80272938762380
// MI455X (gfx1250) — matrix operand order repaired and checked by reading, not compiled, not run on hardware
//
#include <hip/hip_runtime.h>

// Problem dims
#define S_LEN 512
#define BATCH 128
#define EDIM  1024
#define HDIM  1024
#define H4    4096
#define KDIM  2048   // E + H concatenated reduction
#define KC    128    // K chunk staged in LDS per iteration
#define NCOL  64     // 4 gates x 16 columns handled per workgroup

typedef __attribute__((ext_vector_type(16))) __bf16 v16bf;
typedef __attribute__((ext_vector_type(8)))  float  v8f;

union FragU { unsigned int u[8]; uint4 q[2]; v16bf v; };

static __device__ __forceinline__ unsigned short f32_to_bf16(float f) {
    union { float f; unsigned int u; } x; x.f = f;
    unsigned int r = x.u + 0x7FFFu + ((x.u >> 16) & 1u);   // round-to-nearest-even
    return (unsigned short)(r >> 16);
}
static __device__ __forceinline__ unsigned int pack_bf16(float a, float b) {
    return (unsigned int)f32_to_bf16(a) | ((unsigned int)f32_to_bf16(b) << 16);
}
static __device__ __forceinline__ float sigmoidf(float x) {
    return 1.0f / (1.0f + __expf(-x));
}

// --- CDNA5 async global->LDS copy (ASYNCcnt path), guarded fallback ---------
#if defined(__HIP_DEVICE_COMPILE__) && __has_builtin(__builtin_amdgcn_global_load_async_to_lds_b128)
#define HAVE_ASYNC_LDS 1
#else
#define HAVE_ASYNC_LDS 0
#endif

typedef __attribute__((__vector_size__(4 * sizeof(int)))) int v4i_t;
typedef __attribute__((address_space(1))) v4i_t* gv4i_p;   // global (AS1)
typedef __attribute__((address_space(3))) v4i_t* lv4i_p;   // LDS (AS3)

static __device__ __forceinline__ void copy16_g2lds(void* lds_dst, const void* gsrc) {
#if HAVE_ASYNC_LDS
    __builtin_amdgcn_global_load_async_to_lds_b128(
        (gv4i_p)(void*)gsrc,   // generic->AS1 addrspacecast (points to global)
        (lv4i_p)lds_dst,       // generic->AS3 addrspacecast (points to LDS)
        0, 0);
#else
    *(uint4*)lds_dst = *(const uint4*)gsrc;
#endif
}
static __device__ __forceinline__ void wait_async_copies() {
#if HAVE_ASYNC_LDS
#if __has_builtin(__builtin_amdgcn_s_wait_asynccnt)
    __builtin_amdgcn_s_wait_asynccnt(0);
#else
    asm volatile("s_wait_asynccnt 0x0" ::: "memory");
#endif
#endif
}

// ---------------------------------------------------------------------------
// Prep: pack [W_ih ; W_hh] transposed + block-swizzled into bf16:
//   Wpk[g][c][k],  g = column group 0..63, c = gate*16 + n (0..63), k = 0..2047
//   value = k < 1024 ? W_ih[N][k] : W_hh[N][k-1024],  N = (c>>4)*1024 + g*16 + (c&15)
// Column-major per group => each lane's B-fragment is two contiguous 8-bf16 runs.
// ---------------------------------------------------------------------------
__global__ void prep_weights_kernel(const float* __restrict__ W_ih,
                                    const float* __restrict__ W_hh,
                                    unsigned short* __restrict__ Wpk) {
    size_t idx = (size_t)blockIdx.x * blockDim.x + threadIdx.x;
    if (idx >= (size_t)NCOL * 64 * KDIM) return;
    int k = (int)(idx % KDIM);
    int c = (int)((idx / KDIM) % NCOL);
    int g = (int)(idx / ((size_t)KDIM * NCOL));
    int N = ((c >> 4) << 10) + (g << 4) + (c & 15);
    float v = (k < EDIM) ? W_ih[(size_t)N * EDIM + k]
                         : W_hh[(size_t)N * HDIM + (k - EDIM)];
    Wpk[idx] = f32_to_bf16(v);
}

__global__ void prep_bias_kernel(const float* __restrict__ b_ih,
                                 const float* __restrict__ b_hh,
                                 float* __restrict__ bias) {
    int i = blockIdx.x * blockDim.x + threadIdx.x;
    if (i < H4) bias[i] = b_ih[i] + b_hh[i];
}

__global__ void init_state_kernel(unsigned short* __restrict__ h0,
                                  float* __restrict__ c) {
    int i = blockIdx.x * blockDim.x + threadIdx.x;
    if (i < BATCH * HDIM) { h0[i] = 0; c[i] = 0.0f; }
}

// ---------------------------------------------------------------------------
// One LSTM timestep:
//   gates[b, j*H + n] = sum_k [emb(x_t)|h_prev][b,k] * W[k, j*H+n] + bias
//   c_new = sig(f)*c + sig(i)*tanh(g); h_new = sig(o)*tanh(c_new)
// Grid: 64 WGs (one 16-wide column group across all 4 gates).
// Block: 256 threads = 8 waves; wave w owns batch rows [16w, 16w+16).
// ---------------------------------------------------------------------------
__global__ __launch_bounds__(256) void lstm_step_kernel(
    const int*            __restrict__ X,      // [S, B]
    const float*          __restrict__ embd,   // [V, E]
    const unsigned short* __restrict__ Wpk,    // [64][64][K] bf16 swizzled
    const float*          __restrict__ bias,   // [4H]
    const unsigned short* __restrict__ h_in,   // [B, H] bf16
    unsigned short*       __restrict__ h_out,  // [B, H] bf16
    float*                __restrict__ c_st,   // [B, H]
    float*                __restrict__ h_f32,  // [B, H]
    int t)
{
    __shared__ unsigned short As[BATCH][KC + 8];   // activations [row][kk]
    __shared__ unsigned short BsT[NCOL][KC + 8];   // weights, col-major [c][kk]

    const int tid  = threadIdx.x;
    const int wave = tid >> 5;
    const int lane = tid & 31;
    const int g    = blockIdx.x;     // column group 0..63
    const int n0   = g << 4;         // base column inside each gate block
    const int r0   = wave << 4;      // base batch row for this wave
    const int M    = lane & 15;
    const int hi   = lane >> 4;

    v8f acc[4] = {};                 // one f32 16x16 accumulator per gate

    const int* Xt = X + (size_t)t * BATCH;
    const unsigned short* Wg = Wpk + (size_t)g * NCOL * KDIM;

    for (int k0 = 0; k0 < KDIM; k0 += KC) {
        // ---- stage B chunk: 64 cols x 128 k (contiguous per column).
        //      4 threads per column, 64 B each -> 4x b128 async copies.
        {
            const int c = tid >> 2;
            const int q = (tid & 3) << 5;            // 0,32,64,96 (bf16 units)
            const unsigned short* src = Wg + (size_t)c * KDIM + k0 + q;
            unsigned short* dst = &BsT[c][q];
            #pragma unroll
            for (int i = 0; i < 4; ++i)
                copy16_g2lds(dst + 8 * i, src + 8 * i);
            if (k0 + KC < KDIM)                      // next chunk toward L2/L0
                __builtin_prefetch(src + KC, 0, 1);
        }
        // ---- stage A chunk: 128 rows x 128 k. 2 threads/row, 64 elems each.
        {
            const int b   = tid >> 1;
            const int kk0 = (tid & 1) << 6;          // 0 or 64
            const int k   = k0 + kk0;
            if (k0 < EDIM) {
                // gather embedding row, convert f32 -> bf16 (VALU pack)
                const float4* s4 =
                    (const float4*)(embd + (size_t)Xt[b] * EDIM + k);
                unsigned int* d2 = (unsigned int*)&As[b][kk0];
                #pragma unroll
                for (int i = 0; i < 16; ++i) {
                    float4 f = s4[i];
                    d2[2 * i + 0] = pack_bf16(f.x, f.y);
                    d2[2 * i + 1] = pack_bf16(f.z, f.w);
                }
            } else {
                // recurrent h is already bf16: straight async copies
                const unsigned short* src = h_in + (size_t)b * HDIM + (k - EDIM);
                unsigned short* dst = &As[b][kk0];
                #pragma unroll
                for (int i = 0; i < 8; ++i)
                    copy16_g2lds(dst + 8 * i, src + 8 * i);
            }
        }
        wait_async_copies();
        __syncthreads();

        // ---- compute: 4 x (16x16x32) bf16 WMMAs per gate over this chunk
        #pragma unroll
        for (int ks = 0; ks < KC; ks += 32) {
            // A fragment, CDNA5 16-bit A 16x32 layout: per lane two
            // contiguous 8-bf16 runs at K = ks+hi*8 and K = ks+16+hi*8.
            FragU af;
            const unsigned short* arow = &As[r0 + M][0];
            af.q[0] = *(const uint4*)(arow + ks + (hi << 3));
            af.q[1] = *(const uint4*)(arow + ks + 16 + (hi << 3));
            #pragma unroll
            for (int j = 0; j < 4; ++j) {
                // B fragment, 32x16 bf16: lane holds two 8-bf16 runs at
                // column c = j*16 + M, K = ks+hi*8 and K = ks+16+hi*8.
                FragU bf;
                const unsigned short* bcol = &BsT[(j << 4) + M][0];
                const int kb = ks + (hi << 3);
                bf.q[0] = *(const uint4*)(bcol + kb);
                bf.q[1] = *(const uint4*)(bcol + kb + 16);
                acc[j] = __builtin_amdgcn_wmma_f32_16x16x32_bf16(
                    false, af.v, false, bf.v, (short)0, acc[j], false, false);
            }
        }
        __syncthreads();
    }

    // ---- epilogue: gates -> c/h update.  C/D layout: VGPR i holds
    //      row = i (lanes 0-15) or i+8 (lanes 16-31), col = lane&15.
    const int col = n0 + M;
    const float bi  = bias[0 * HDIM + col];
    const float bff = bias[1 * HDIM + col];
    const float bg  = bias[2 * HDIM + col];
    const float bo  = bias[3 * HDIM + col];
    #pragma unroll
    for (int i = 0; i < 8; ++i) {
        const int row = r0 + i + (hi << 3);
        const size_t idx = (size_t)row * HDIM + col;
        const float gi = sigmoidf(acc[0][i] + bi);
        const float gf = sigmoidf(acc[1][i] + bff);
        const float gg = tanhf(acc[2][i] + bg);
        const float go = sigmoidf(acc[3][i] + bo);
        const float cn = gf * c_st[idx] + gi * gg;
        c_st[idx] = cn;
        const float hn = go * tanhf(cn);
        h_f32[idx] = hn;
        h_out[idx] = f32_to_bf16(hn);
    }
}

// ---------------------------------------------------------------------------
// log_softmax over the BATCH axis (axis=1 of [1,B,H]) — per hidden column
// ---------------------------------------------------------------------------
__global__ void logsoftmax_batch_kernel(const float* __restrict__ h,
                                        float* __restrict__ out) {
    int col = blockIdx.x * blockDim.x + threadIdx.x;
    if (col >= HDIM) return;
    float m = -INFINITY;
    for (int b = 0; b < BATCH; ++b) m = fmaxf(m, h[(size_t)b * HDIM + col]);
    float s = 0.0f;
    for (int b = 0; b < BATCH; ++b) s += __expf(h[(size_t)b * HDIM + col] - m);
    const float lse = m + __logf(s);
    for (int b = 0; b < BATCH; ++b)
        out[(size_t)b * HDIM + col] = h[(size_t)b * HDIM + col] - lse;
}

// ---------------------------------------------------------------------------
extern "C" void kernel_launch(void* const* d_in, const int* in_sizes, int n_in,
                              void* d_out, int out_size, void* d_ws, size_t ws_size,
                              hipStream_t stream) {
    (void)in_sizes; (void)n_in; (void)out_size; (void)ws_size;
    const int*   X     = (const int*)  d_in[0];
    const float* embd  = (const float*)d_in[1];
    const float* W_ih  = (const float*)d_in[2];
    const float* W_hh  = (const float*)d_in[3];
    const float* b_ih  = (const float*)d_in[4];
    const float* b_hh  = (const float*)d_in[5];
    float* out = (float*)d_out;

    // workspace carve-out (~18 MB)
    char* ws = (char*)d_ws;
    unsigned short* Wpk  = (unsigned short*)ws; ws += (size_t)KDIM * H4 * 2;  // 16 MB
    float*          bias = (float*)ws;          ws += (size_t)H4 * 4;
    unsigned short* h0   = (unsigned short*)ws; ws += (size_t)BATCH * HDIM * 2;
    unsigned short* h1   = (unsigned short*)ws; ws += (size_t)BATCH * HDIM * 2;
    float*          c    = (float*)ws;          ws += (size_t)BATCH * HDIM * 4;
    float*          hf   = (float*)ws;

    prep_weights_kernel<<<(KDIM * H4 + 255) / 256, 256, 0, stream>>>(W_ih, W_hh, Wpk);
    prep_bias_kernel<<<(H4 + 255) / 256, 256, 0, stream>>>(b_ih, b_hh, bias);
    init_state_kernel<<<(BATCH * HDIM + 255) / 256, 256, 0, stream>>>(h0, c);

    unsigned short* hin  = h0;
    unsigned short* hout = h1;
    for (int t = 0; t < S_LEN; ++t) {
        lstm_step_kernel<<<HDIM / 16, 256, 0, stream>>>(
            X, embd, Wpk, bias, hin, hout, c, hf, t);
        unsigned short* tmp = hin; hin = hout; hout = tmp;
    }
    logsoftmax_batch_kernel<<<(HDIM + 255) / 256, 256, 0, stream>>>(hf, out);
}
